// SPFNet_A_56599079026974
// MI455X (gfx1250) — matrix operand order repaired and checked by reading, not compiled, not run on hardware
//
#include <hip/hip_runtime.h>
#include <hip/hip_bf16.h>

// ---------------------------------------------------------------------------
// SPFNet on MI455X (gfx1250): wave32, 16x16x32 f16 WMMA (f32 accum) GEMMs fed
// by async global->LDS tile copies (ASYNCcnt path), L2-resident edge gathers,
// f32 atomics for segment reductions.
// ---------------------------------------------------------------------------

typedef _Float16 f16x16 __attribute__((ext_vector_type(16)));
typedef _Float16 f16x8  __attribute__((ext_vector_type(8)));
typedef float    f32x8  __attribute__((ext_vector_type(8)));
typedef int      i32x4  __attribute__((ext_vector_type(4)));

#if __has_builtin(__builtin_amdgcn_global_load_async_to_lds_b128)
#define USE_ASYNC_LDS 1
#else
#define USE_ASYNC_LDS 0
#endif

#define AS1 __attribute__((address_space(1)))
#define AS3 __attribute__((address_space(3)))

// ======================= generic fills =====================================
__global__ void fill_f32(float* p, float v, long n) {
  long i = (long)blockIdx.x * blockDim.x + threadIdx.x;
  long st = (long)gridDim.x * blockDim.x;
  for (; i < n; i += st) p[i] = v;
}
__global__ void fill_u32_k(unsigned* p, unsigned v, long n) {
  long i = (long)blockIdx.x * blockDim.x + threadIdx.x;
  long st = (long)gridDim.x * blockDim.x;
  for (; i < n; i += st) p[i] = v;
}

// ============ f32 -> f16 convert with zero-padded row pitch ================
__global__ void cvt_f16_pad(const float* __restrict__ x, _Float16* __restrict__ o,
                            int M, int K, int Kpad) {
  long n = (long)M * Kpad;
  long i = (long)blockIdx.x * blockDim.x + threadIdx.x;
  long st = (long)gridDim.x * blockDim.x;
  for (; i < n; i += st) {
    long r = i / Kpad;
    int kp = (int)(i - r * Kpad);
    o[i] = (kp < K) ? (_Float16)x[r * K + kp] : (_Float16)0.f;
  }
}

// ======================= WMMA GEMM: out[M,128] = A[M,K] @ W + bias =========
// A is pre-converted f16, row pitch Kpad (multiple of 32, zero padded).
// Weight addressing: w(k,c) = W[k*wks + (c>>whs)*whstride + (c & wcm)]
//   normal 128-col weight : wks=128, whs=7, whstride=0,      wcm=127
//   2-head (2x[128,64])   : wks=64,  whs=6, whstride=128*64, wcm=63
__global__ __launch_bounds__(256) void gemm_wmma_f16(
    const _Float16* __restrict__ Ah, int Kpad, int M, int K,
    const float* __restrict__ W, int wks, int whs, int whstride, int wcm,
    const float* __restrict__ bias, float* __restrict__ out)
{
  __shared__ alignas(16) _Float16 Al[128 * 40];  // [row][k], 40-half (80B) pitch
  __shared__ alignas(16) _Float16 Bl[128 * 40];  // [col][k], 40-half pitch

  const int tid  = threadIdx.x;
  const int wave = tid >> 5;
  const int lane = tid & 31;
  const int sub  = lane & 15;
  const int hi   = lane >> 4;
  const int rowBase = blockIdx.x * 128;

  f32x8 acc[8];
#pragma unroll
  for (int s = 0; s < 8; ++s)
#pragma unroll
    for (int j = 0; j < 8; ++j) acc[s][j] = 0.0f;

  for (int k0 = 0; k0 < Kpad; k0 += 32) {
    // ---- stage A tile: 128 rows x 32 halves (64B/row) -> LDS -------------
    // 512 x 16B chunks, 2 per thread; async copy engine when available.
#pragma unroll
    for (int i = 0; i < 2; ++i) {
      int chunk = tid + i * 256;
      int r = chunk >> 2, c16 = (chunk & 3) * 8;             // 8 halves = 16B
      const _Float16* gp = Ah + (size_t)(rowBase + r) * Kpad + k0 + c16;
      if (k0 + 32 < Kpad) __builtin_prefetch(gp + 32, 0, 1); // global_prefetch_b8
#if USE_ASYNC_LDS
      __builtin_amdgcn_global_load_async_to_lds_b128(
          (AS1 i32x4*)gp, (AS3 i32x4*)&Al[r * 40 + c16], 0, 0);
#else
      *(f16x8*)&Al[r * 40 + c16] = *(const f16x8*)gp;
#endif
    }
    // ---- stage B tile: 32 k x 128 cols (f32 -> f16, transposed [c][k]) ---
#pragma unroll
    for (int i = 0; i < 4; ++i) {
      int slot = tid + i * 256;                 // 1024 slots = 32 k * 32 col-quads
      int kk = slot >> 5, c4 = (slot & 31) * 4;
      int gk = k0 + kk;
      float4 v = make_float4(0.f, 0.f, 0.f, 0.f);
      if (gk < K) {
        size_t wi = (size_t)gk * wks + (size_t)(c4 >> whs) * whstride + (c4 & wcm);
        v = *(const float4*)(W + wi);
      }
      Bl[(c4 + 0) * 40 + kk] = (_Float16)v.x;
      Bl[(c4 + 1) * 40 + kk] = (_Float16)v.y;
      Bl[(c4 + 2) * 40 + kk] = (_Float16)v.z;
      Bl[(c4 + 3) * 40 + kk] = (_Float16)v.w;
    }
#if USE_ASYNC_LDS
#if __has_builtin(__builtin_amdgcn_s_wait_asynccnt)
    __builtin_amdgcn_s_wait_asynccnt(0);
#else
    asm volatile("s_wait_asynccnt 0" ::: "memory");
#endif
#endif
    __syncthreads();

    // ---- A fragment (16x32): lane<16 -> K{0..7,16..23}; lane>=16 -> K{8..15,24..31}
    int ar = wave * 16 + sub;
    f16x8 a0 = *reinterpret_cast<const f16x8*>(&Al[ar * 40 + hi * 8]);
    f16x8 a1 = *reinterpret_cast<const f16x8*>(&Al[ar * 40 + 16 + hi * 8]);
    f16x16 af;
#pragma unroll
    for (int i = 0; i < 8; ++i) { af[i] = a0[i]; af[i + 8] = a1[i]; }

#pragma unroll
    for (int s = 0; s < 8; ++s) {
      // ---- B fragment (32x16): col = 16s+sub, lane<16 -> K{0..7,16..23}; lane>=16 -> K{8..15,24..31}
      int bc = s * 16 + sub;
      f16x8 b0 = *reinterpret_cast<const f16x8*>(&Bl[bc * 40 + hi * 8]);
      f16x8 b1 = *reinterpret_cast<const f16x8*>(&Bl[bc * 40 + 16 + hi * 8]);
      f16x16 bf;
#pragma unroll
      for (int i = 0; i < 8; ++i) { bf[i] = b0[i]; bf[i + 8] = b1[i]; }
      acc[s] = __builtin_amdgcn_wmma_f32_16x16x32_f16(
          false, af, false, bf, (short)0, acc[s], false, false);
    }
    __syncthreads();
  }

  // ---- epilogue: C/D layout -> row = base + j + 8*hi, col = 16s + sub
#pragma unroll
  for (int s = 0; s < 8; ++s) {
    int col = s * 16 + sub;
    float bb = bias ? bias[col] : 0.0f;
#pragma unroll
    for (int j = 0; j < 8; ++j) {
      int grow = rowBase + wave * 16 + hi * 8 + j;
      if (grow < M) out[(size_t)grow * 128 + col] = acc[s][j] + bb;
    }
  }
}

// ======================= BatchNorm (training stats) ========================
__global__ __launch_bounds__(256) void col_stats(const float* __restrict__ x,
                                                 float* __restrict__ stats, int M) {
  __shared__ float ssum[128], ssq[128];
  int tid = threadIdx.x;
  if (tid < 128) { ssum[tid] = 0.f; ssq[tid] = 0.f; }
  __syncthreads();
  int c = tid & 127;
  int ph = tid >> 7;
  float s = 0.f, q = 0.f;
  for (int r = blockIdx.x * 2 + ph; r < M; r += gridDim.x * 2) {
    float v = x[(size_t)r * 128 + c];
    s += v; q += v * v;
  }
  atomicAdd(&ssum[c], s);
  atomicAdd(&ssq[c], q);
  __syncthreads();
  if (tid < 128) {
    atomicAdd(&stats[tid], ssum[tid]);
    atomicAdd(&stats[128 + tid], ssq[tid]);
  }
}

__global__ void bn_apply(const float* __restrict__ x, const float* __restrict__ stats,
                         const float* __restrict__ g, const float* __restrict__ b,
                         float* __restrict__ o, int M, int lrelu) {
  long n = (long)M * 128;
  long i = (long)blockIdx.x * blockDim.x + threadIdx.x;
  long st = (long)gridDim.x * blockDim.x;
  float invM = 1.0f / (float)M;
  for (; i < n; i += st) {
    int c = (int)(i & 127);
    float mean = stats[c] * invM;
    float var  = fmaxf(stats[128 + c] * invM - mean * mean, 0.f);
    float y = (x[i] - mean) * rsqrtf(var + 1e-5f) * g[c] + b[c];
    if (lrelu) y = (y >= 0.f) ? y : 0.01f * y;
    o[i] = y;
  }
}

// ======================= superpixel pooling ================================
__global__ void seg_pool(const float* __restrict__ x, const int* __restrict__ seg,
                         float* __restrict__ acc, float* __restrict__ cnt, int N) {
  long idx = (long)blockIdx.x * blockDim.x + threadIdx.x;
  if (idx >= (long)N * 32) return;
  int p = (int)(idx >> 5), cg = (int)(idx & 31), c = cg * 4;
  int s = seg[p];
  float4 v = *(const float4*)&x[(size_t)p * 128 + c];
  float* d = &acc[(size_t)s * 128 + c];
  atomicAdd(d + 0, v.x); atomicAdd(d + 1, v.y);
  atomicAdd(d + 2, v.z); atomicAdd(d + 3, v.w);
  if (cg == 0) atomicAdd(&cnt[s], 1.0f);
}

__global__ void seg_div(const float* __restrict__ acc, const float* __restrict__ cnt,
                        float* __restrict__ o, long n) {
  long i = (long)blockIdx.x * blockDim.x + threadIdx.x;
  if (i >= n) return;
  o[i] = acc[i] / fmaxf(cnt[i >> 7], 1.0f);
}

// ======================= SFNet propagate: o[dst] += w * x[src] =============
__global__ void sf_prop(const float* __restrict__ x, const int* __restrict__ src,
                        const int* __restrict__ dst, const float* __restrict__ w,
                        float* __restrict__ o, int E) {
  long idx = (long)blockIdx.x * blockDim.x + threadIdx.x;
  if (idx >= (long)E * 32) return;
  int e = (int)(idx >> 5), c = (int)(idx & 31) * 4;
  int s = src[e], d = dst[e];
  float ww = w[e];
  float4 v = *(const float4*)&x[(size_t)s * 128 + c];
  float* op = &o[(size_t)d * 128 + c];
  atomicAdd(op + 0, ww * v.x); atomicAdd(op + 1, ww * v.y);
  atomicAdd(op + 2, ww * v.z); atomicAdd(op + 3, ww * v.w);
}

__global__ void combine_inplace(float* __restrict__ p, const float* __restrict__ xs,
                                float inv, long n) {
  long i = (long)blockIdx.x * blockDim.x + threadIdx.x;
  if (i >= n) return;
  p[i] = (p[i] + xs[i]) * inv;
}

// ======================= PFNet: per-head LayerNorm (no affine) =============
__global__ void ln_heads(float* __restrict__ q, int N) {
  long idx = (long)blockIdx.x * blockDim.x + threadIdx.x;
  if (idx >= (long)N * 2) return;
  float* p = q + (size_t)(idx >> 1) * 128 + (idx & 1) * 64;
  float s = 0.f, ss = 0.f;
#pragma unroll
  for (int k = 0; k < 64; k += 4) {
    float4 v = *(const float4*)&p[k];
    s  += v.x + v.y + v.z + v.w;
    ss += v.x * v.x + v.y * v.y + v.z * v.z + v.w * v.w;
  }
  float m = s * (1.0f / 64.0f);
  float var = fmaxf(ss * (1.0f / 64.0f) - m * m, 0.f);
  float r = rsqrtf(var + 1e-5f);
#pragma unroll
  for (int k = 0; k < 64; ++k) p[k] = (p[k] - m) * r;
}

// monotonic float<->u32 encoding for atomicMax-based segment max
__device__ __forceinline__ unsigned fenc(float f) {
  int i = __float_as_int(f);
  return (i >= 0) ? ((unsigned)i | 0x80000000u) : ~(unsigned)i;
}
__device__ __forceinline__ float fdec(unsigned k) {
  int i = (k & 0x80000000u) ? (int)(k & 0x7FFFFFFFu) : (int)(~k);
  return __int_as_float(i);
}

__global__ void edge_scores(const float* __restrict__ q, const int* __restrict__ src,
                            const int* __restrict__ dst, float* __restrict__ sb, long E) {
  long idx = (long)blockIdx.x * blockDim.x + threadIdx.x;
  if (idx >= E * 2) return;
  long e = idx >> 1;
  int h = (int)(idx & 1);
  int a = src[e], bN = dst[e];
  const float* qa = q + (size_t)a * 128 + h * 64;
  const float* qb = q + (size_t)bN * 128 + h * 64;
  float acc = 0.f;
#pragma unroll
  for (int k = 0; k < 64; k += 4) {
    float4 va = *(const float4*)&qa[k];
    float4 vb = *(const float4*)&qb[k];
    acc += va.x * vb.x + va.y * vb.y + va.z * vb.z + va.w * vb.w;
  }
  sb[idx] = acc * (1.0f / 128.0f);   // /HID
}

__global__ void edge_max(const float* __restrict__ sb, const int* __restrict__ src,
                         unsigned* __restrict__ mx, long E) {
  long idx = (long)blockIdx.x * blockDim.x + threadIdx.x;
  if (idx >= E * 2) return;
  long e = idx >> 1;
  int h = (int)(idx & 1);
  atomicMax(&mx[(size_t)src[e] * 2 + h], fenc(sb[idx]));
}

__global__ void edge_expsum(float* __restrict__ sb, const int* __restrict__ src,
                            const unsigned* __restrict__ mx, float* __restrict__ z, long E) {
  long idx = (long)blockIdx.x * blockDim.x + threadIdx.x;
  if (idx >= E * 2) return;
  long e = idx >> 1;
  int h = (int)(idx & 1);
  int s = src[e];
  float ev = __expf(sb[idx] - fdec(mx[(size_t)s * 2 + h]));
  sb[idx] = ev;
  atomicAdd(&z[(size_t)s * 2 + h], ev);
}

__global__ void edge_agg(const float* __restrict__ sb, const float* __restrict__ z,
                         const float* __restrict__ v, const int* __restrict__ src,
                         const int* __restrict__ dst, float* __restrict__ o, long E) {
  long idx = (long)blockIdx.x * blockDim.x + threadIdx.x;
  if (idx >= E * 32) return;
  long e = idx >> 5;
  int cg = (int)(idx & 31);
  int head = cg >> 4;
  int c = cg * 4;
  int s = src[e], d = dst[e];
  float att = sb[e * 2 + head] / z[(size_t)s * 2 + head];
  float4 vv = *(const float4*)&v[(size_t)d * 128 + c];
  float* op = &o[(size_t)s * 128 + c];
  atomicAdd(op + 0, att * vv.x); atomicAdd(op + 1, att * vv.y);
  atomicAdd(op + 2, att * vv.z); atomicAdd(op + 3, att * vv.w);
}

// ======================= final: softmax((pf + hp[seg]) @ W + b) ============
__global__ __launch_bounds__(256) void final_softmax(
    const float* __restrict__ pf, const float* __restrict__ hp,
    const int* __restrict__ seg, const float* __restrict__ W,
    const float* __restrict__ b, float* __restrict__ out, int N) {
  __shared__ float Wl[128 * 16];
  __shared__ float bl[16];
  for (int i = threadIdx.x; i < 2048; i += 256) Wl[i] = W[i];
  if (threadIdx.x < 16) bl[threadIdx.x] = b[threadIdx.x];
  __syncthreads();
  int row = blockIdx.x * blockDim.x + threadIdx.x;
  if (row >= N) return;
  const float* pr = pf + (size_t)row * 128;
  const float* hr = hp + (size_t)seg[row] * 128;
  float acc[16];
#pragma unroll
  for (int c = 0; c < 16; ++c) acc[c] = bl[c];
  for (int k = 0; k < 128; ++k) {
    float h = pr[k] + hr[k];
#pragma unroll
    for (int c = 0; c < 16; ++c) acc[c] = fmaf(h, Wl[k * 16 + c], acc[c]);
  }
  float mx = acc[0];
#pragma unroll
  for (int c = 1; c < 16; ++c) mx = fmaxf(mx, acc[c]);
  float sum = 0.f;
#pragma unroll
  for (int c = 0; c < 16; ++c) { acc[c] = __expf(acc[c] - mx); sum += acc[c]; }
  float inv = 1.0f / sum;
#pragma unroll
  for (int c = 0; c < 16; ++c) out[(size_t)row * 16 + c] = acc[c] * inv;
}

// ===========================================================================
extern "C" void kernel_launch(void* const* d_in, const int* in_sizes, int n_in,
                              void* d_out, int out_size, void* d_ws, size_t ws_size,
                              hipStream_t stream) {
  (void)in_sizes; (void)n_in; (void)out_size; (void)ws_size;
  constexpr int N = 262144, C = 200, HID = 128, S = 4096, EA = 65536;
  constexpr long EM = 2097152;
  constexpr int CPAD = 224;   // 200 rounded up to 32

  const float* x     = (const float*)d_in[0];
  const int*   seg   = (const int*)d_in[1];
  const int*   a_src = (const int*)d_in[2];
  const int*   a_dst = (const int*)d_in[3];
  const float* a_w   = (const float*)d_in[4];
  const int*   m_src = (const int*)d_in[5];
  const int*   m_dst = (const int*)d_in[6];
  const float* pre_W = (const float*)d_in[7];
  const float* pre_b = (const float*)d_in[8];
  const float* bn0_g = (const float*)d_in[9];
  const float* bn0_b = (const float*)d_in[10];
  const float* sf_W  = (const float*)d_in[11];
  const float* sf_b  = (const float*)d_in[12];
  const float* sf_g  = (const float*)d_in[13];
  const float* sf_be = (const float*)d_in[14];
  const float* pf_Wv = (const float*)d_in[15];
  const float* pf_bv = (const float*)d_in[16];
  const float* pf_Wq = (const float*)d_in[17];
  const float* pf_bq = (const float*)d_in[18];
  const float* pf_g  = (const float*)d_in[19];
  const float* pf_be = (const float*)d_in[20];
  const float* out_W = (const float*)d_in[21];
  const float* out_b = (const float*)d_in[22];
  float* out = (float*)d_out;

  // ---- workspace carve-up -------------------------------------------------
  float* ws = (float*)d_ws;
  size_t off = 0;
  auto take = [&](size_t n) { float* p = ws + off; off += n; return p; };
  float* X0  = take((size_t)N * HID);    // x0 / pf scratch
  float* V   = take((size_t)N * HID);    // v / pf out
  float* Q   = take((size_t)N * HID);    // q / pf out
  float* AGG = take((size_t)N * HID);    // attention aggregate
  float* EB  = take((size_t)EM * 2);     // edge scores / exp
  float* Z   = take((size_t)N * 2);      // softmax denominators
  unsigned* MX = (unsigned*)take((size_t)N * 2);  // encoded segment max
  float* HP  = take((size_t)S * HID);    // superpixel features
  float* HB  = take((size_t)S * HID);    // pool-sum / sf GEMM out
  float* XST = take((size_t)S * HID);    // x_start
  float* P0  = take((size_t)S * HID);
  float* P1  = take((size_t)S * HID);
  float* CNT = take(S);
  float* STATS = take(256);
  _Float16* AH = (_Float16*)take((size_t)N * CPAD / 2);  // padded f16 GEMM input

  auto fillf = [&](float* p, float v, long n) {
    long blocks = (n + 255) / 256; if (blocks > 8192) blocks = 8192;
    fill_f32<<<dim3((unsigned)blocks), dim3(256), 0, stream>>>(p, v, n);
  };
  auto cvt = [&](const float* src, int M, int K, int Kpad) {
    long n = (long)M * Kpad;
    long blocks = (n + 255) / 256; if (blocks > 8192) blocks = 8192;
    cvt_f16_pad<<<dim3((unsigned)blocks), dim3(256), 0, stream>>>(src, AH, M, K, Kpad);
  };
  auto gemm = [&](int Kpad, int M, int K, const float* W,
                  int wks, int whs, int whstride, int wcm,
                  const float* bias, float* o) {
    gemm_wmma_f16<<<dim3(M / 128), dim3(256), 0, stream>>>(
        AH, Kpad, M, K, W, wks, whs, whstride, wcm, bias, o);
  };
  auto bnrun = [&](const float* src, const float* g, const float* b,
                   float* dst, int M, int lrelu) {
    fillf(STATS, 0.f, 256);
    col_stats<<<dim3(512), dim3(256), 0, stream>>>(src, STATS, M);
    bn_apply<<<dim3(4096), dim3(256), 0, stream>>>(src, STATS, g, b, dst, M, lrelu);
  };

  // ---- 1) x0 = BN(x @ pre_W + pre_b) -------------------------------------
  cvt(x, N, C, CPAD);
  gemm(CPAD, N, C, pre_W, 128, 7, 0, 127, pre_b, X0);
  bnrun(X0, bn0_g, bn0_b, X0, N, 0);

  // ---- 2) superpixel mean pooling ----------------------------------------
  fillf(HB, 0.f, (long)S * HID);
  fillf(CNT, 0.f, S);
  seg_pool<<<dim3((N * 32) / 256), dim3(256), 0, stream>>>(X0, seg, HB, CNT, N);
  seg_div<<<dim3((S * 128) / 256), dim3(256), 0, stream>>>(HB, CNT, HP, (long)S * 128);

  // ---- 3) SFNet: 5 layers on superpixel graph ----------------------------
  const float inv29 = 1.0f / 2.9f;   // 1/(2+GAMA)
  for (int l = 0; l < 5; ++l) {
    cvt(HP, S, 128, 128);
    gemm(128, S, 128, sf_W + (size_t)l * 128 * 128, 128, 7, 0, 127, sf_b + l * 128, HB);
    fillf(XST, 0.f, (long)S * HID);
    sf_prop<<<dim3((EA * 32) / 256), dim3(256), 0, stream>>>(HB, a_src, a_dst, a_w, XST, EA);
    const float* cur = HP;
    for (int t = 0; t < 5; ++t) {
      float* nxt = (t & 1) ? P1 : P0;
      fillf(nxt, 0.f, (long)S * HID);
      sf_prop<<<dim3((EA * 32) / 256), dim3(256), 0, stream>>>(cur, a_src, a_dst, a_w, nxt, EA);
      combine_inplace<<<dim3((S * 128) / 256), dim3(256), 0, stream>>>(nxt, XST, inv29,
                                                                       (long)S * 128);
      cur = nxt;
    }
    bnrun(cur, sf_g + l * 128, sf_be + l * 128, HP, S, 1);
  }

  // ---- 4) PFNet: 2 attention layers on pixel graph -----------------------
  const float* Xin = X0;
  const float* pfFinal = nullptr;
  for (int l = 0; l < 2; ++l) {
    float* Vb = (l == 0) ? V : X0;   // value buffer (prev buffers recycled)
    float* Qb = (l == 0) ? Q : V;    // query buffer; BN output reuses it
    const size_t wo = (size_t)l * 2 * 128 * 64;
    cvt(Xin, N, 128, 128);           // shared f16 input for both GEMMs
    gemm(128, N, 128, pf_Wv + wo, 64, 6, 128 * 64, 63, pf_bv + l * 128, Vb);
    gemm(128, N, 128, pf_Wq + wo, 64, 6, 128 * 64, 63, pf_bq + l * 128, Qb);
    ln_heads<<<dim3((N * 2) / 256), dim3(256), 0, stream>>>(Qb, N);
    edge_scores<<<dim3((unsigned)((EM * 2) / 256)), dim3(256), 0, stream>>>(Qb, m_src,
                                                                            m_dst, EB, EM);
    fill_u32_k<<<dim3(8192), dim3(256), 0, stream>>>(MX, 0x007FFFFFu, (long)N * 2); // enc(-inf)
    edge_max<<<dim3((unsigned)((EM * 2) / 256)), dim3(256), 0, stream>>>(EB, m_src, MX, EM);
    fillf(Z, 0.f, (long)N * 2);
    edge_expsum<<<dim3((unsigned)((EM * 2) / 256)), dim3(256), 0, stream>>>(EB, m_src, MX,
                                                                            Z, EM);
    fillf(AGG, 0.f, (long)N * HID);
    edge_agg<<<dim3((unsigned)((EM * 32) / 256)), dim3(256), 0, stream>>>(EB, Z, Vb, m_src,
                                                                          m_dst, AGG, EM);
    bnrun(AGG, pf_g + l * 128, pf_be + l * 128, Qb, N, 1);
    Xin = Qb;
    pfFinal = Qb;
  }

  // ---- 5) output head -----------------------------------------------------
  final_softmax<<<dim3(N / 256), dim3(256), 0, stream>>>(pfFinal, HP, seg, out_W, out_b,
                                                         out, N);
}
